// CNNBlock_3770981285925
// MI455X (gfx1250) — matrix operand order repaired and checked by reading, not compiled, not run on hardware
//
#include <hip/hip_runtime.h>
#include <hip/hip_bf16.h>
#include <math.h>

// MI455X / gfx1250 fused implementation.
//
// Roofline: ~182 GFLOP total vs ~27MB unique bytes -> compute bound by >100x.
// Dominant cost: ff = relu(x@Bw^T+Bb) @ Cw^T  (4096 x 128 x 8128) x 20 steps.
// => v_wmma_f32_16x16x32_bf16 (bf16 A/B, f32 accum); weights pre-converted to
//    bf16 in d_ws. Key layout fact: the B-operand LDS slab [n][k] = B[k][n]
//    equals the *row-major source matrix itself* for x@A^T, x@Bw^T, x@W1^T and
//    for Cw chunks, so staging is a plain 2D tile copy -> done with the
//    Tensor Data Mover (tensor_load_to_lds), using TDM's pad feature
//    (pad 16B per 256B row) to produce the bank-padded 272B LDS row stride.
// One 256-thread workgroup (8 wave32) owns a 16-row batch tile persistently in
// LDS; ff [16][8192] bf16 = 256KB exploits CDNA5's 320KB/WGP LDS.

#define NFD   128
#define BMD   16
#define CHD   512
#define LD    4
#define SD    5
#define NSKW  8128
#define NPADP 8192          // pairs padded to 64 chunks of 128
#define WSTR  136           // slab row stride in bf16 (256B + 16B pad = 272B)
#define WSL_OFF 24576u      // LDS byte offset of the weight slab

typedef __attribute__((ext_vector_type(16))) __bf16 v16bf;
typedef __attribute__((ext_vector_type(8)))  float  v8f;

union BFV { v16bf v; unsigned short s[16]; };

__device__ __forceinline__ unsigned short f2bf(float f) {
  union { float f; unsigned int u; } x; x.f = f;
  unsigned int r = x.u + 0x7FFFu + ((x.u >> 16) & 1u);   // RNE
  return (unsigned short)(r >> 16);
}
__device__ __forceinline__ float bf2f(unsigned short h) {
  union { unsigned int u; float f; } x; x.u = ((unsigned int)h) << 16;
  return x.f;
}

// A operand (16x32 bf16, M x K): lane m=l&15, half=l>>4.
// elements 0..7  = K[kb + half*16 .. +7]   (consecutive -> ds_load_b128)
// elements 8..15 = K[kb + half*16 + 8 .. +7]
__device__ __forceinline__ v16bf load_a16(const unsigned short* buf, int stride,
                                          int m, int half, int kb) {
  BFV r;
  const unsigned short* p0 = buf + m * stride + kb + half * 16;
  #pragma unroll
  for (int i = 0; i < 8; i++) r.s[i] = p0[i];
  const unsigned short* p1 = p0 + 8;
  #pragma unroll
  for (int i = 0; i < 8; i++) r.s[8 + i] = p1[i];
  return r.v;
}

// B operand (32x16 bf16, K x N): lane n=l&15, K = half*16 + i.
// slab stored [n][k] so the 16 elements are consecutive (2x ds_load_b128).
__device__ __forceinline__ v16bf load_b16(const unsigned short* wsl,
                                          int col, int half, int kb) {
  BFV r;
  const unsigned short* p = wsl + col * WSTR + kb + half * 16;
  #pragma unroll
  for (int i = 0; i < 16; i++) r.s[i] = p[i];
  return r.v;
}

// One 16x16 output tile over K=128 (4 chained wmma).
__device__ __forceinline__ v8f gemm_tile128(const unsigned short* Abuf, int astride,
                                            const unsigned short* wsl,
                                            int lane, int col16, v8f c) {
  const int m = lane & 15, half = lane >> 4;
  const int col = col16 + m;
  #pragma unroll
  for (int kt = 0; kt < 4; kt++) {
    v16bf a = load_a16(Abuf, astride, m, half, kt * 32);
    v16bf b = load_b16(wsl, col, half, kt * 32);
    c = __builtin_amdgcn_wmma_f32_16x16x32_bf16(false, a, false, b,
                                                (short)0, c, false, false);
  }
  return c;
}

#define HAS_TDM __has_builtin(__builtin_amdgcn_tensor_load_to_lds)

#if HAS_TDM
typedef __attribute__((ext_vector_type(4))) unsigned int v4u;
typedef __attribute__((ext_vector_type(8))) int v8i32;
typedef __attribute__((ext_vector_type(4))) int v4i32;

// TDM: async-copy a 128(row) x 128(elem) bf16 tile (row stride = strideElems)
// from global into LDS at ldsByteOff, inserting 16B pad after each 256B row
// => LDS row stride 272B == WSTR*2. Issued by one wave; TENSORcnt tracked.
__device__ __forceinline__ void tdm_stage(const unsigned short* src, int strideElems,
                                          unsigned int ldsByteOff) {
  unsigned long long ga = (unsigned long long)(size_t)src;
  v4u g0;
  g0[0] = 1u;                                    // count=1, load, not restore
  g0[1] = ldsByteOff;                            // lds_addr
  g0[2] = (unsigned int)ga;                      // global_addr[31:0]
  g0[3] = (unsigned int)(ga >> 32) | (2u << 30); // global_addr[56:32] | type=2
  v8i32 g1;
  g1[0] = (int)0x07510000;       // data_size=2B | pad_en | interval=64dw | amount=4dw
  g1[1] = (int)(128u << 16);     // tensor_dim0 = 128
  g1[2] = (int)(128u << 16);     // tensor_dim1 = 128
  g1[3] = (int)(128u << 16);     // tile_dim0 = 128
  g1[4] = (int)128;              // tile_dim1 = 128 (tile_dim2 = 0)
  g1[5] = strideElems;           // tensor_dim0_stride (data_size units)
  g1[6] = 0;
  g1[7] = 0;
  v4i32 z4 = {0, 0, 0, 0};
  v8i32 z8 = {0, 0, 0, 0, 0, 0, 0, 0};
  // clang-23 / therock lane: 6-arg form (g0, g1, g2, g3, extra, cpol)
  __builtin_amdgcn_tensor_load_to_lds(g0, g1, z4, z4, z8, 0);
}
#endif

// Fallback cooperative stage: contiguous reads AND contiguous writes.
__device__ __forceinline__ void stage_slab(const unsigned short* __restrict__ src,
                                           int srcStride,
                                           unsigned short* __restrict__ wsl,
                                           int tid) {
  #pragma unroll
  for (int r = 0; r < 16; r++) {
    int q = (r * 256 + tid) * 4;          // 0..16383 in steps of 4
    int n = q >> 7, k = q & 127;
    const unsigned short* sp = src + n * srcStride + k;
    unsigned short v0 = sp[0], v1 = sp[1], v2 = sp[2], v3 = sp[3];
    unsigned short* dp = wsl + n * WSTR + k;
    dp[0] = v0; dp[1] = v1; dp[2] = v2; dp[3] = v3;
  }
}

// stage helper used inside the kernel (TDM if available, else cooperative)
#if HAS_TDM
#define STAGE(srcp, stridee) do {                                   \
    if (wave == 0) {                                                \
      tdm_stage((srcp), (stridee), WSL_OFF);                        \
      __builtin_amdgcn_s_wait_tensorcnt(0);                         \
    }                                                               \
  } while (0)
#else
#define STAGE(srcp, stridee) stage_slab((srcp), (stridee), wsl, tid)
#endif

// ---------------- prep kernels: fp32 weights -> bf16 slab-ready layouts -----
__global__ __launch_bounds__(256)
void prep_small_k(const float* __restrict__ W1, const float* __restrict__ A,
                  const float* __restrict__ Bw, const float* __restrict__ u,
                  float* __restrict__ dtp,
                  unsigned short* __restrict__ W1B, unsigned short* __restrict__ AB,
                  unsigned short* __restrict__ ATB, unsigned short* __restrict__ BWB) {
  int gid = blockIdx.x * 256 + threadIdx.x;            // 65536 threads
  if (gid < LD) dtp[gid] = log1pf(expf(u[gid])) * (1.0f / (float)SD);
  W1B[gid] = f2bf(W1[gid]);                            // direct (slab = W1 rows)
  {
    int l = gid >> 14, w = gid & 16383, n = w >> 7, k = w & 127;
    const float* Al = A + l * NFD * NFD;
    AB[gid]  = f2bf(Al[w]);                // direct   (B-op for x@A^T)
    ATB[gid] = f2bf(Al[k * NFD + n]);      // transpose (B-op for h@A)
    BWB[gid] = f2bf(Bw[l * NFD * NFD + w]);// direct   (B-op for x@Bw^T)
  }
}

__global__ __launch_bounds__(256)
void prep_cw_k(const float* __restrict__ Cw, unsigned short* __restrict__ CWB) {
  size_t gid = (size_t)blockIdx.x * 256 + threadIdx.x;  // 4 * 2^20 threads
  int l = (int)(gid >> 20);
  int w = (int)(gid & 1048575);
  int p = w >> 7;                // padded pair row
  float v = 0.f;
  if (p < NSKW) v = Cw[(size_t)l * (NSKW * NFD) + w];   // direct rows of Cw
  CWB[gid] = f2bf(v);
}

// ---------------- fused main kernel: 16 batch rows per workgroup ------------
__global__ __launch_bounds__(256)
void fused_net_k(const float* __restrict__ X,  const float* __restrict__ b1,
                 const float* __restrict__ bA, const float* __restrict__ alpha,
                 const float* __restrict__ Bb, const float* __restrict__ dtp,
                 const unsigned short* __restrict__ W1B,
                 const unsigned short* __restrict__ AB,
                 const unsigned short* __restrict__ ATB,
                 const unsigned short* __restrict__ BWB,
                 const unsigned short* __restrict__ CWB,
                 float* __restrict__ out) {
  extern __shared__ unsigned char smem[];
  float*          xs   = (float*)(smem);                   //  8 KB  x tile f32
  float*          ys   = (float*)(smem + 8192);            //  8 KB  x_next
  unsigned short* xb   = (unsigned short*)(smem + 16384);  //  4 KB  x bf16
  unsigned short* hs   = (unsigned short*)(smem + 20480);  //  4 KB  h bf16
  unsigned short* wsl  = (unsigned short*)(smem + WSL_OFF);// 34 KB  weight slab
  unsigned short* ff   = (unsigned short*)(smem + 59392);  // 256 KB ff bf16
  unsigned short* xbig = ff;                               // phase-0 overlay

  const int tid  = threadIdx.x;
  const int lane = tid & 31;
  const int wave = tid >> 5;
  const int m15  = lane & 15;
  const int half = lane >> 4;
  const int mb   = half * 8;                 // C-tile row base for this lane
  const int ncol = wave * 16 + m15;          // output column this lane owns
  (void)wave;

  // ===== phase 0: x = X @ W1^T + b1   (K = 512, 4 slabs) =====
  {
    const float* Xt = X + (size_t)blockIdx.x * BMD * CHD;
    #pragma unroll
    for (int r = 0; r < 32; r++) { int e = r * 256 + tid; xbig[e] = f2bf(Xt[e]); }
  }
  __syncthreads();
  v8f acc = {};
  #pragma unroll 1
  for (int ks = 0; ks < 4; ks++) {
    STAGE(W1B + ks * 128, CHD);
    __syncthreads();
    acc = gemm_tile128(xbig + ks * 128, CHD, wsl, lane, wave * 16, acc);
    __syncthreads();
  }
  {
    float bias = b1[ncol];
    #pragma unroll
    for (int r = 0; r < 8; r++) xs[(mb + r) * NFD + ncol] = acc[r] + bias;
  }
  __syncthreads();

  #pragma unroll 1
  for (int li = 0; li < LD; li++) {
    const float dtv = dtp[li];
    const float* bAi = bA + li * NFD;
    const float* ali = alpha + li * NFD;
    const float* Bbi = Bb + li * NFD;
    const unsigned short* ABi  = AB  + li * NFD * NFD;
    const unsigned short* ATBi = ATB + li * NFD * NFD;
    const unsigned short* BWBi = BWB + li * NFD * NFD;
    const unsigned short* CWBi = CWB + (size_t)li * NPADP * NFD;

    // ===== gradient-flow steps: x += dt * (relu(x@A^T + b) @ A) * alpha =====
    #pragma unroll 1
    for (int s = 0; s < SD; s++) {
      #pragma unroll
      for (int r = 0; r < 8; r++) { int e = r * 256 + tid; xb[e] = f2bf(xs[e]); }
      STAGE(ABi, NFD);
      __syncthreads();
      v8f hc = {};
      hc = gemm_tile128(xb, NFD, wsl, lane, wave * 16, hc);
      {
        float bias = bAi[ncol];
        #pragma unroll
        for (int r = 0; r < 8; r++) {
          float v = hc[r] + bias; v = v > 0.f ? v : 0.f;
          hs[(mb + r) * NFD + ncol] = f2bf(v);
        }
      }
      __syncthreads();
      STAGE(ATBi, NFD);
      __syncthreads();
      v8f gc = {};
      gc = gemm_tile128(hs, NFD, wsl, lane, wave * 16, gc);
      {
        float av = ali[ncol];
        #pragma unroll
        for (int r = 0; r < 8; r++)
          xs[(mb + r) * NFD + ncol] += dtv * gc[r] * av;
      }
      __syncthreads();
    }

    // ===== skew-field steps =====
    #pragma unroll 1
    for (int s = 0; s < SD; s++) {
      #pragma unroll
      for (int r = 0; r < 8; r++) { int e = r * 256 + tid; xb[e] = f2bf(xs[e]); }
      STAGE(BWBi, NFD);
      __syncthreads();
      v8f hc = {};
      hc = gemm_tile128(xb, NFD, wsl, lane, wave * 16, hc);
      {
        float bias = Bbi[ncol];
        #pragma unroll
        for (int r = 0; r < 8; r++) {
          float v = hc[r] + bias; v = v > 0.f ? v : 0.f;
          hs[(mb + r) * NFD + ncol] = f2bf(v);
        }
      }
      __syncthreads();

      // ff[16][8192] = h @ Cw^T : 64 chunks, each chunk slab = 128 rows of Cw
      #pragma unroll 1
      for (int c = 0; c < 64; c++) {
        STAGE(CWBi + (size_t)c * 128 * NFD, NFD);
        __syncthreads();
        v8f fc = {};
        fc = gemm_tile128(hs, NFD, wsl, lane, wave * 16, fc);
        const int pc = c * 128 + ncol;
        #pragma unroll
        for (int r = 0; r < 8; r++) ff[(mb + r) * NPADP + pc] = f2bf(fc[r]);
        __syncthreads();
      }

      // y[b,j] = sum_{k>j} ff[b,off(j)+k-j-1]*x[b,k] - sum_{i<j} ff[b,off(i)+j-i-1]*x[b,i]
      // (contiguous run for the first sum, stride 126-i for the second); no atomics.
      {
        const int j   = tid & 127;
        const int bb0 = (tid >> 7) * 8;
        const int baseU = j * 127 - (j * (j - 1)) / 2;
        #pragma unroll 1
        for (int bb = 0; bb < 8; bb++) {
          const int b = bb0 + bb;
          const unsigned short* ffb = ff + b * NPADP;
          const float* xr = xs + b * NFD;
          float y = 0.f;
          for (int k = j + 1; k < NFD; k++)
            y += bf2f(ffb[baseU + (k - j - 1)]) * xr[k];
          int idx = j - 1;
          for (int i = 0; i < j; i++) { y -= bf2f(ffb[idx]) * xr[i]; idx += 126 - i; }
          ys[b * NFD + j] = xr[j] + dtv * y;
        }
      }
      __syncthreads();
      #pragma unroll
      for (int r = 0; r < 8; r++) { int e = r * 256 + tid; xs[e] = ys[e]; }
      __syncthreads();
    }
  }

  // ===== store result tile =====
  {
    float* op = out + (size_t)blockIdx.x * BMD * NFD;
    #pragma unroll
    for (int r = 0; r < 8; r++) { int e = r * 256 + tid; op[e] = xs[e]; }
  }
}

// ---------------- host launch ----------------
extern "C" void kernel_launch(void* const* d_in, const int* in_sizes, int n_in,
                              void* d_out, int out_size, void* d_ws, size_t ws_size,
                              hipStream_t stream) {
  const float* X     = (const float*)d_in[0];
  const float* W1    = (const float*)d_in[1];
  const float* b1    = (const float*)d_in[2];
  const float* A     = (const float*)d_in[3];
  const float* bA    = (const float*)d_in[4];
  const float* alpha = (const float*)d_in[5];
  const float* Bw    = (const float*)d_in[6];
  const float* Bb    = (const float*)d_in[7];
  const float* Cw    = (const float*)d_in[8];
  const float* u     = (const float*)d_in[9];
  float* out = (float*)d_out;

  // workspace layout (bf16 weight copies + dt): ~8.9 MB
  char* ws = (char*)d_ws;
  float*          dtp = (float*)ws;                                  // 4 f32
  unsigned short* W1B = (unsigned short*)(ws + 256);                 // 128x512
  unsigned short* AB  = (unsigned short*)(ws + 256 + 1 * 131072);    // 4x128x128
  unsigned short* ATB = (unsigned short*)(ws + 256 + 2 * 131072);
  unsigned short* BWB = (unsigned short*)(ws + 256 + 3 * 131072);
  unsigned short* CWB = (unsigned short*)(ws + 256 + 4 * 131072);    // 4x8192x128

  prep_small_k<<<dim3(256), dim3(256), 0, stream>>>(W1, A, Bw, u, dtp, W1B, AB, ATB, BWB);
  prep_cw_k<<<dim3(16384), dim3(256), 0, stream>>>(Cw, CWB);

  const size_t lds_bytes = 59392 + 16 * NPADP * 2;  // 321,536 B (<= 320KB WGP LDS)
  fused_net_k<<<dim3(4096 / BMD), dim3(256), lds_bytes, stream>>>(
      X, b1, bA, alpha, Bb, dtp, W1B, AB, ATB, BWB, CWB, out);
}
